// SelfAttention_91156385890836
// MI455X (gfx1250) — matrix operand order repaired and checked by reading, not compiled, not run on hardware
//
#include <hip/hip_runtime.h>
#include <hip/hip_bf16.h>
#include <cstdint>

// ---------------- problem constants ----------------
constexpr int B  = 8;
constexpr int S  = 2048;
constexpr int E  = 128;
constexpr int CC = 256;
constexpr int H  = 256;
constexpr int NH = 8;
constexpr int HD = 32;   // head dim
constexpr int KW = 7;    // conv kernel width
constexpr int PAD = 8;   // padded halo rows for conv (need KW/2=3, rounded up)
constexpr int SP = S + 2 * PAD;

typedef _Float16 half8  __attribute__((ext_vector_type(8)));
typedef _Float16 v16h   __attribute__((ext_vector_type(16)));
typedef float    v8f    __attribute__((ext_vector_type(8)));

// ---------------- WMMA fragment helpers (CDNA5 ISA 7.12.2) ----------------
// A 16x32 f16: lane L (M = L&15). lanes<16 hold K 0..7 & 16..23; lanes>=16 hold K 8..15 & 24..31.
__device__ __forceinline__ v16h frag_a(const _Float16* tile, int ld, int lane) {
  const int m  = lane & 15;
  const int hi = (lane >> 4) & 1;
  const _Float16* p = tile + (size_t)m * ld + hi * 8;
  half8 lo = *(const half8*)(p);
  half8 up = *(const half8*)(p + 16);
  v16h r;
#pragma unroll
  for (int i = 0; i < 8; ++i) { r[i] = lo[i]; r[i + 8] = up[i]; }
  return r;
}
// B 32x16 f16 from a transposed (N-major, K-contiguous) buffer: lane L holds column N=L&15,
// lanes<16 hold K 0..7 & 16..23; lanes>=16 hold K 8..15 & 24..31 (matches frag_a).
__device__ __forceinline__ v16h frag_bT(const _Float16* tileT, int ld, int lane) {
  const int n  = lane & 15;
  const int hi = (lane >> 4) & 1;
  const _Float16* p = tileT + (size_t)n * ld + hi * 8;
  half8 lo = *(const half8*)(p);
  half8 up = *(const half8*)(p + 16);
  v16h r;
#pragma unroll
  for (int i = 0; i < 8; ++i) { r[i] = lo[i]; r[i + 8] = up[i]; }
  return r;
}
__device__ __forceinline__ v8f acc_zero() {
  v8f a;
#pragma unroll
  for (int i = 0; i < 8; ++i) a[i] = 0.f;
  return a;
}
__device__ __forceinline__ v8f wmma_f16(v16h a, v16h b, v8f c) {
  return __builtin_amdgcn_wmma_f32_16x16x32_f16(false, a, false, b, (short)0, c, false, false);
}
// per-wave LDS ordering: LDS ops of one wave execute in order at the LDS unit;
// this just drains DScnt and stops the compiler reordering across it.
__device__ __forceinline__ void lds_fence_wave() {
  asm volatile("s_wait_dscnt 0x0" ::: "memory");
}

// ---------------- conversion kernels ----------------
// x (B,S,E) f32 -> zero-padded f16 (B, S+2*PAD, E): halo rows are zero so the conv
// kernel needs no bounds checks (straight-line b128 loads, no exec-mask branches).
__global__ void cvt_x_pad(const float* __restrict__ x, _Float16* __restrict__ xp) {
  int i = blockIdx.x * blockDim.x + threadIdx.x;
  if (i >= B * SP * E) return;
  int e = i % E; int t = i / E; int srow = t % SP; int b = t / SP;
  int s = srow - PAD;
  xp[i] = (s >= 0 && s < S) ? (_Float16)x[((size_t)b * S + s) * E + e] : (_Float16)0.f;
}
// conv_w [CC][E][KW] -> wconvT [k][cc][e]  (B_k transposed: out-major, in-contiguous)
__global__ void cvt_convw(const float* __restrict__ w, _Float16* __restrict__ wT) {
  int i = blockIdx.x * blockDim.x + threadIdx.x;  // i = (k*CC+cc)*E + e
  if (i >= KW * CC * E) return;
  int e = i % E; int t = i / E; int cc = t % CC; int k = t / CC;
  wT[i] = (_Float16)w[(cc * E + e) * KW + k];
}
// 256x256 [in][out] -> [out][in], optionally scaled (folds 1/sqrt(HD) into wq)
__global__ void cvt_sq_T(const float* __restrict__ w, _Float16* __restrict__ wT, float scale) {
  int i = blockIdx.x * blockDim.x + threadIdx.x;  // i = o*256 + in
  if (i >= 256 * 256) return;
  int in = i & 255; int o = i >> 8;
  wT[i] = (_Float16)(w[in * 256 + o] * scale);
}

// ---------------- penalty vector + lengths ----------------
__global__ void len_pvec_kernel(const unsigned char* __restrict__ mask,
                                const float* __restrict__ gamma_p,
                                const float* __restrict__ std_p,
                                int* __restrict__ len_out, float* __restrict__ pvec) {
  int b = blockIdx.x;
  int t = threadIdx.x;
  __shared__ int partial[256];
  int cnt = 0;
  for (int s = t; s < S; s += 256) cnt += (mask[b * S + s] != 0) ? 1 : 0;
  partial[t] = cnt;
  __syncthreads();
  for (int off = 128; off > 0; off >>= 1) { if (t < off) partial[t] += partial[t + off]; __syncthreads(); }
  __shared__ int len_s;
  if (t == 0) { len_s = partial[0]; len_out[b] = len_s; }
  __syncthreads();
  int len = len_s;
  float vlen = (float)len;
  float stdc = fminf(fmaxf(std_p[0], 0.01f), 0.5f);
  float egm1 = __expf(gamma_p[0]) - 1.f;
  for (int s = t; s < S; s += 256) {
    float np = (s + 1.0f) / (vlen + 1.0f);
    float d  = np - 0.5f;
    float base = __expf(-(d * d) / (2.f * stdc * stdc));
    pvec[b * S + s] = (s < len) ? base * egm1 : 0.f;
  }
}

// ---------------- conv1d as 7 shifted GEMMs over the padded buffer ----------------
__global__ void conv_kernel(const _Float16* __restrict__ xp, const _Float16* __restrict__ wconvT,
                            const float* __restrict__ conv_b,
                            float* __restrict__ hf, _Float16* __restrict__ hh) {
  int gw   = blockIdx.x * 4 + (threadIdx.x >> 5);
  int lane = threadIdx.x & 31;
  int rt = gw >> 4, ct = gw & 15;               // 1024 row tiles x 16 col tiles
  int b  = rt / (S / 16);
  int s0 = (rt % (S / 16)) * 16;
  int cc0 = ct * 16;
  int m  = lane & 15;
  int hi = (lane >> 4) & 1;
  v8f acc = acc_zero();
  // lane's A row for all taps: padded row index s0 + m + PAD + (k - KW/2), always in bounds
  const _Float16* xrow0 = xp + ((size_t)b * SP + (s0 + m + PAD - KW / 2)) * E + hi * 8;
#pragma unroll
  for (int k = 0; k < KW; ++k) {
    const _Float16* xrow = xrow0 + (size_t)k * E;
#pragma unroll
    for (int ec = 0; ec < E / 32; ++ec) {
      const _Float16* p = xrow + ec * 32;
      half8 lo = *(const half8*)(p);
      half8 up = *(const half8*)(p + 16);
      v16h a;
#pragma unroll
      for (int i = 0; i < 8; ++i) { a[i] = lo[i]; a[i + 8] = up[i]; }
      v16h bf = frag_bT(wconvT + ((size_t)k * CC + cc0) * E + ec * 32, E, lane);
      acc = wmma_f16(a, bf, acc);
    }
  }
  int cc = cc0 + m;
  float bias = conv_b[cc];
#pragma unroll
  for (int j = 0; j < 8; ++j) {
    int s = s0 + j + hi * 8;
    float v = fmaxf(acc[j] + bias, 0.f);
    size_t idx = ((size_t)b * S + s) * CC + cc;
    hf[idx] = v;
    hh[idx] = (_Float16)v;
  }
}

// ---------------- QKV projections ----------------
__global__ void qkv_kernel(const _Float16* __restrict__ hh,
                           const _Float16* __restrict__ wqT, const _Float16* __restrict__ wkT,
                           const _Float16* __restrict__ wvT,
                           const float* __restrict__ bq, const float* __restrict__ bk,
                           const float* __restrict__ bv,
                           _Float16* __restrict__ q, _Float16* __restrict__ kmat,
                           _Float16* __restrict__ vT) {
  int gw   = blockIdx.x * 4 + (threadIdx.x >> 5);
  int lane = threadIdx.x & 31;
  const int TILES = (B * S / 16) * (H / 16);     // 16384 per matrix
  int which = gw / TILES;
  int r = gw % TILES;
  int rt = r >> 4, ct = r & 15;
  int b  = rt / (S / 16);
  int s0 = (rt % (S / 16)) * 16;
  int h0 = ct * 16;
  const _Float16* wT = (which == 0) ? wqT : (which == 1) ? wkT : wvT;
  const float* bias  = (which == 0) ? bq  : (which == 1) ? bk  : bv;
  v8f acc = acc_zero();
  const _Float16* arow = hh + (size_t)(rt * 16) * CC;
#pragma unroll
  for (int kc = 0; kc < CC / 32; ++kc) {
    v16h a  = frag_a(arow + kc * 32, CC, lane);
    v16h bf = frag_bT(wT + (size_t)h0 * CC + kc * 32, CC, lane);
    acc = wmma_f16(a, bf, acc);
  }
  int m = lane & 15, hi = (lane >> 4) & 1;
  int h = h0 + m;
  int nh = h >> 5, d = h & 31;
  const float qscale = 0.17677669529663687f;  // 1/sqrt(HD); weights pre-scaled, bias is not
  float bb = (which == 0) ? bias[h] * qscale : bias[h];
#pragma unroll
  for (int j = 0; j < 8; ++j) {
    int s = s0 + j + hi * 8;
    float v = acc[j] + bb;
    if (which == 0) {
      q[(((size_t)b * NH + nh) * S + s) * HD + d] = (_Float16)v;
    } else if (which == 1) {
      kmat[(((size_t)b * NH + nh) * S + s) * HD + d] = (_Float16)v;
    } else {
      vT[(((size_t)b * NH + nh) * HD + d) * S + s] = (_Float16)v;
    }
  }
}

// ---------------- fused flash attention (penalty + mask + online softmax) ----------------
__global__ void attn_kernel(const _Float16* __restrict__ q, const _Float16* __restrict__ kmat,
                            const _Float16* __restrict__ vT,
                            const float* __restrict__ pvec, const int* __restrict__ len_arr,
                            _Float16* __restrict__ ctx) {
  __shared__ _Float16 pl[4][16 * 32];   // per-wave P tile (D-layout -> A-layout transpose)
  int wid  = threadIdx.x >> 5;
  int lane = threadIdx.x & 31;
  int gw   = blockIdx.x * 4 + wid;      // 0..8191
  int qt = gw % (S / 16);
  int bn = gw / (S / 16);
  int nh = bn % NH;
  int b  = bn / NH;
  int s0 = qt * 16;
  int len = len_arr[b];
  int m = lane & 15, hi = (lane >> 4) & 1;

  const _Float16* qbase = q    + (((size_t)b * NH + nh) * S + s0) * HD;
  const _Float16* kb_   = kmat + ((size_t)b * NH + nh) * S * HD;
  const _Float16* vb_   = vT   + ((size_t)b * NH + nh) * HD * S;
  _Float16* pbuf = pl[wid];

  v16h qa = frag_a(qbase, HD, lane);

  float rm[8], rl[8], pv_row[8], rowv[8];
  v8f o0 = acc_zero(), o1 = acc_zero();
#pragma unroll
  for (int j = 0; j < 8; ++j) {
    rm[j] = -3.0e38f; rl[j] = 0.f;
    int qi = s0 + j + hi * 8;
    rowv[j]   = (qi < len) ? 1.f : 0.f;
    pv_row[j] = (qi < len) ? pvec[b * S + qi] : 0.f;
  }

  for (int kc = 0; kc < S; kc += 32) {
    // prefetch next K chunk into cache (gfx1250 global_prefetch_b8)
    if (kc + 32 < S) __builtin_prefetch(kb_ + (size_t)(kc + 32) * HD + lane * 16, 0, 0);

    // scores: q(16x32) @ K^T(32 x 32keys) -> two 16x16 D tiles
    v16h kf0 = frag_bT(kb_ + (size_t)kc * HD,        HD, lane);
    v16h kf1 = frag_bT(kb_ + (size_t)(kc + 16) * HD, HD, lane);
    v8f sa = wmma_f16(qa, kf0, acc_zero());
    v8f sb = wmma_f16(qa, kf1, acc_zero());

    int n0 = kc + m, n1 = n0 + 16;
    float pv0 = (n0 < len) ? pvec[b * S + n0] : 0.f;
    float pv1 = (n1 < len) ? pvec[b * S + n1] : 0.f;

#pragma unroll
    for (int j = 0; j < 8; ++j) {
      float sc0 = sa[j] - 0.5f * (pv_row[j] + pv0) * rowv[j];
      float sc1 = sb[j] - 0.5f * (pv_row[j] + pv1) * rowv[j];
      if (n0 >= len) sc0 = -1.0e10f;
      if (n1 >= len) sc1 = -1.0e10f;
      // row max across the 16 lanes holding this row (xor masks 1..8 stay in half)
      float cm = fmaxf(sc0, sc1);
#pragma unroll
      for (int off = 8; off > 0; off >>= 1) cm = fmaxf(cm, __shfl_xor(cm, off, 32));
      float nm = fmaxf(rm[j], cm);
      float p0 = __expf(sc0 - nm);
      float p1 = __expf(sc1 - nm);
      float sm = p0 + p1;
#pragma unroll
      for (int off = 8; off > 0; off >>= 1) sm += __shfl_xor(sm, off, 32);
      float scale = __expf(rm[j] - nm);
      rl[j] = rl[j] * scale + sm;
      rm[j] = nm;
      o0[j] *= scale;
      o1[j] *= scale;
      int row = j + hi * 8;
      pbuf[row * 32 + m]      = (_Float16)p0;
      pbuf[row * 32 + 16 + m] = (_Float16)p1;
    }
    // per-wave private P tile: LDS unit processes one wave's DS ops in order,
    // so a DScnt drain (not a workgroup barrier) is all that's needed.
    lds_fence_wave();

    v16h pa  = frag_a(pbuf, 32, lane);
    // P(16x32keys) @ V(32keys x 32d): B-frags from vT (d-major, key-contiguous)
    v16h vf0 = frag_bT(vb_ + 0 * (size_t)S + kc,  S, lane);   // d 0..15
    v16h vf1 = frag_bT(vb_ + 16 * (size_t)S + kc, S, lane);   // d 16..31
    o0 = wmma_f16(pa, vf0, o0);
    o1 = wmma_f16(pa, vf1, o1);
    lds_fence_wave();  // keep this iteration's reads ordered before next iteration's writes
  }

#pragma unroll
  for (int j = 0; j < 8; ++j) {
    int s = s0 + j + hi * 8;
    float inv = (rl[j] > 0.f) ? (1.f / rl[j]) : 0.f;
    size_t base = ((size_t)b * S + s) * H + nh * 32;
    ctx[base + m]      = (_Float16)(o0[j] * inv);
    ctx[base + 16 + m] = (_Float16)(o1[j] * inv);
  }
}

// ---------------- output projection + residual ----------------
__global__ void outproj_kernel(const _Float16* __restrict__ ctx, const _Float16* __restrict__ fcT,
                               const float* __restrict__ fc_b, const float* __restrict__ hf,
                               float* __restrict__ z) {
  int gw   = blockIdx.x * 4 + (threadIdx.x >> 5);
  int lane = threadIdx.x & 31;
  int rt = gw >> 4, ct = gw & 15;
  int cc0 = ct * 16;
  v8f acc = acc_zero();
  const _Float16* arow = ctx + (size_t)(rt * 16) * H;
#pragma unroll
  for (int kc = 0; kc < H / 32; ++kc) {
    v16h a  = frag_a(arow + kc * 32, H, lane);
    v16h bf = frag_bT(fcT + (size_t)cc0 * H + kc * 32, H, lane);
    acc = wmma_f16(a, bf, acc);
  }
  int m = lane & 15, hi = (lane >> 4) & 1;
  int cc = cc0 + m;
  float bias = fc_b[cc];
#pragma unroll
  for (int j = 0; j < 8; ++j) {
    int row = rt * 16 + j + hi * 8;
    size_t idx = (size_t)row * CC + cc;
    z[idx] = hf[idx] + acc[j] + bias;
  }
}

// ---------------- row LayerNorm (in place over d_out) ----------------
__global__ void ln_kernel(float* __restrict__ z, const float* __restrict__ g,
                          const float* __restrict__ beta) {
  int row  = blockIdx.x * 8 + (threadIdx.x >> 5);
  int lane = threadIdx.x & 31;
  float* zr = z + (size_t)row * CC;
  float v[8];
  float s = 0.f, s2 = 0.f;
#pragma unroll
  for (int i = 0; i < 8; ++i) {
    v[i] = zr[lane * 8 + i];
    s += v[i]; s2 += v[i] * v[i];
  }
#pragma unroll
  for (int off = 16; off > 0; off >>= 1) {
    s  += __shfl_xor(s, off, 32);
    s2 += __shfl_xor(s2, off, 32);
  }
  float mu  = s * (1.f / CC);
  float var = s2 * (1.f / CC) - mu * mu;
  float inv = rsqrtf(var + 1e-5f);
#pragma unroll
  for (int i = 0; i < 8; ++i) {
    int c = lane * 8 + i;
    zr[c] = (v[i] - mu) * inv * g[c] + beta[c];
  }
}

// ---------------- host launcher ----------------
extern "C" void kernel_launch(void* const* d_in, const int* in_sizes, int n_in,
                              void* d_out, int out_size, void* d_ws, size_t ws_size,
                              hipStream_t stream) {
  (void)in_sizes; (void)n_in; (void)out_size; (void)ws_size;
  const float* x      = (const float*)d_in[0];
  const unsigned char* mask = (const unsigned char*)d_in[1];
  const float* conv_w = (const float*)d_in[2];
  const float* conv_b = (const float*)d_in[3];
  const float* wq     = (const float*)d_in[4];
  const float* bq     = (const float*)d_in[5];
  const float* wk     = (const float*)d_in[6];
  const float* bk     = (const float*)d_in[7];
  const float* wv     = (const float*)d_in[8];
  const float* bv     = (const float*)d_in[9];
  const float* fcw    = (const float*)d_in[10];
  const float* fcb    = (const float*)d_in[11];
  const float* lng    = (const float*)d_in[12];
  const float* lnb    = (const float*)d_in[13];
  const float* gamma  = (const float*)d_in[14];
  const float* stdp   = (const float*)d_in[15];
  float* out = (float*)d_out;

  uintptr_t base = (uintptr_t)d_ws;
  size_t off = 0;
  auto carve = [&](size_t bytes) -> void* {
    void* p = (void*)(base + off);
    off = (off + bytes + 255) & ~(size_t)255;
    return p;
  };
  _Float16* xp     = (_Float16*)carve((size_t)B * SP * E * 2);   // zero-padded x
  _Float16* wconvT = (_Float16*)carve((size_t)KW * CC * E * 2);
  _Float16* wqT    = (_Float16*)carve((size_t)CC * H * 2);
  _Float16* wkT    = (_Float16*)carve((size_t)CC * H * 2);
  _Float16* wvT    = (_Float16*)carve((size_t)CC * H * 2);
  _Float16* fcT    = (_Float16*)carve((size_t)H * CC * 2);
  float*    hf     = (float*)   carve((size_t)B * S * CC * 4);
  _Float16* hh     = (_Float16*)carve((size_t)B * S * CC * 2);
  _Float16* qh     = (_Float16*)carve((size_t)B * S * H * 2);
  _Float16* kh     = (_Float16*)carve((size_t)B * S * H * 2);
  _Float16* vTh    = (_Float16*)carve((size_t)B * S * H * 2);
  _Float16* ctxh   = (_Float16*)carve((size_t)B * S * H * 2);
  float*    pvec   = (float*)   carve((size_t)B * S * 4);
  int*      lenb   = (int*)     carve((size_t)B * 4);

  // conversions
  {
    int n = B * SP * E;
    cvt_x_pad<<<(n + 255) / 256, 256, 0, stream>>>(x, xp);
  }
  cvt_convw<<<(KW * CC * E + 255) / 256, 256, 0, stream>>>(conv_w, wconvT);
  const float qscale = 0.17677669529663687f;  // 1/sqrt(HD)
  cvt_sq_T<<<256, 256, 0, stream>>>(wq, wqT, qscale);
  cvt_sq_T<<<256, 256, 0, stream>>>(wk, wkT, 1.f);
  cvt_sq_T<<<256, 256, 0, stream>>>(wv, wvT, 1.f);
  cvt_sq_T<<<256, 256, 0, stream>>>(fcw, fcT, 1.f);
  len_pvec_kernel<<<B, 256, 0, stream>>>(mask, gamma, stdp, lenb, pvec);

  // conv frontend: (B*S/16)*(CC/16) = 16384 tiles, 4 waves/block
  conv_kernel<<<16384 / 4, 128, 0, stream>>>(xp, wconvT, conv_b, hf, hh);

  // QKV: 3 * 16384 tiles
  qkv_kernel<<<(3 * 16384) / 4, 128, 0, stream>>>(hh, wqT, wkT, wvT, bq, bk, bv, qh, kh, vTh);

  // flash attention: B*NH*(S/16) = 8192 q-tiles
  attn_kernel<<<8192 / 4, 128, 0, stream>>>(qh, kh, vTh, pvec, lenb, ctxh);

  // out projection + residual -> d_out (f32)
  outproj_kernel<<<16384 / 4, 128, 0, stream>>>(ctxh, fcT, fcb, hf, out);

  // layernorm in place over d_out: 16384 rows, 8 waves/block
  ln_kernel<<<16384 / 8, 256, 0, stream>>>(out, lng, lnb);
}
